// ViTWithBiFormer_23184233463881
// MI455X (gfx1250) — matrix operand order repaired and checked by reading, not compiled, not run on hardware
//
#include <hip/hip_runtime.h>
#include <math.h>

// ---------------------------------------------------------------------------
// ViT + BiFormer forward for gfx1250 (MI455X). All GEMMs via
// v_wmma_f32_16x16x32_bf16 (f32 accumulate). b128 global loads, v_perm_b32
// bf16 packing, software-pipelined LDS staging. Compile-only target.
// ---------------------------------------------------------------------------

typedef __attribute__((ext_vector_type(16))) __bf16 v16bf;
typedef __attribute__((ext_vector_type(8)))  float  v8f;

union Frag16 {
    v16bf v;
    unsigned int u[8];
};

// fp32 -> bf16 (round-half-up), 2 VALU ops
__device__ __forceinline__ unsigned short f2bf(float f) {
    union { float f; unsigned int u; } x;
    x.f = f;
    return (unsigned short)((x.u + 0x8000u) >> 16);
}

// pack two fp32 -> bf16x2 dword (lo in [15:0], hi in [31:16]):
// two v_add + one v_perm_b32 selecting the high halves of {hi, lo}
__device__ __forceinline__ unsigned int pk2(float lo, float hi) {
    union { float f; unsigned int u; } a, b;
    a.f = lo; b.f = hi;
    return __builtin_amdgcn_perm(b.u + 0x8000u, a.u + 0x8000u, 0x07060302u);
}

// ---------------------------------------------------------------------------
// Generic bf16 WMMA GEMM:  C[M,N] = act(A[M,K] * B[K,N] + bias) + res
// fp32 row-major A/B/res/C. Requirements (all uses satisfy them):
//   K multiple of 32, N multiple of 4, A/B 16B aligned.
// Workgroup 256 threads = 8 waves; C tile 128x128; wave tile 32x64 (2x4 WMMA).
// 2-stage pipeline: global fetch of tile k+1 overlaps WMMAs of tile k.
// ---------------------------------------------------------------------------
#define BM 128
#define BN 128
#define BK 32
#define LDA 40   // halfs per LDS row: 16B-aligned rows, conflict-spreading

__global__ __launch_bounds__(256)
void gemm_bf16_wmma(const float* __restrict__ A, const float* __restrict__ Bm,
                    const float* __restrict__ bias, const float* __restrict__ res,
                    float* __restrict__ C, int M, int N, int K, int act)
{
    __shared__ unsigned short As[BM][LDA];
    __shared__ unsigned short Bs[BN][LDA];   // transposed: Bs[n][k]

    const int tid  = threadIdx.x;
    const int lane = tid & 31;
    const int w    = tid >> 5;
    const int half = lane >> 4;
    const int l16  = lane & 15;
    const int wm   = w & 3;        // 0..3 -> M offset wm*32
    const int wn   = w >> 2;       // 0..1 -> N offset wn*64
    const int tM   = blockIdx.y * BM;
    const int tN   = blockIdx.x * BN;
    const bool aFull = (tM + BM) <= M;

    // A loader: each thread owns 16 consecutive k of one row (4x float4)
    const int arow = tid >> 1;
    const int akq  = (tid & 1) << 4;       // 0 or 16
    const int agm  = tM + arow;
    const bool aOk = aFull || (agm < M);
    const float* aptr = A + (size_t)agm * K + akq;

    // B loader: each thread owns a 4(k) x 4(n) block
    const int bn4 = (tid >> 3) << 2;       // 0,4,...,124
    const int bk4 = (tid & 7) << 2;        // 0,4,...,28
    const int bgn = tN + bn4;
    const bool bOk = bgn < N;              // N multiple of 4 => full float4 ok
    const float* bptr = Bm + (size_t)bk4 * N + bgn;

    // staged (packed bf16) registers; stay zero for out-of-range threads
    unsigned int aP[8] = {0, 0, 0, 0, 0, 0, 0, 0};
    unsigned int bP[8] = {0, 0, 0, 0, 0, 0, 0, 0};

    auto fetch = [&](int k0) {
        if (aOk) {
            const float4* p = (const float4*)(aptr + k0);
            float4 q0 = p[0], q1 = p[1], q2 = p[2], q3 = p[3];
            aP[0] = pk2(q0.x, q0.y); aP[1] = pk2(q0.z, q0.w);
            aP[2] = pk2(q1.x, q1.y); aP[3] = pk2(q1.z, q1.w);
            aP[4] = pk2(q2.x, q2.y); aP[5] = pk2(q2.z, q2.w);
            aP[6] = pk2(q3.x, q3.y); aP[7] = pk2(q3.z, q3.w);
        }
        if (bOk) {
            const float* bp = bptr + (size_t)k0 * N;
            float4 r0 = *(const float4*)(bp);
            float4 r1 = *(const float4*)(bp + (size_t)N);
            float4 r2 = *(const float4*)(bp + 2 * (size_t)N);
            float4 r3 = *(const float4*)(bp + 3 * (size_t)N);
            bP[0] = pk2(r0.x, r1.x); bP[1] = pk2(r2.x, r3.x);
            bP[2] = pk2(r0.y, r1.y); bP[3] = pk2(r2.y, r3.y);
            bP[4] = pk2(r0.z, r1.z); bP[5] = pk2(r2.z, r3.z);
            bP[6] = pk2(r0.w, r1.w); bP[7] = pk2(r2.w, r3.w);
        }
    };

    v8f acc[2][4] = {};
    fetch(0);

    for (int k0 = 0; k0 < K; k0 += BK) {
        // ---- commit staged tile to LDS ----
        *(uint4*)&As[arow][akq]     = make_uint4(aP[0], aP[1], aP[2], aP[3]);
        *(uint4*)&As[arow][akq + 8] = make_uint4(aP[4], aP[5], aP[6], aP[7]);
        #pragma unroll
        for (int c = 0; c < 4; ++c)
            *(uint2*)&Bs[bn4 + c][bk4] = make_uint2(bP[2 * c], bP[2 * c + 1]);
        __syncthreads();

        // ---- prefetch next tile (overlaps with WMMAs below) ----
        if (k0 + BK < K) fetch(k0 + BK);

        Frag16 af[2], bfr[4];
        #pragma unroll
        for (int i = 0; i < 2; ++i) {
            const int row = wm * 32 + i * 16 + l16;
            const int kb  = half * 16;  // lanes 0-15: K 0..15 ; lanes 16-31: K 16..31
            #pragma unroll
            for (int p = 0; p < 4; ++p) {
                af[i].u[p]     = *(const unsigned int*)&As[row][kb + 2 * p];
                af[i].u[4 + p] = *(const unsigned int*)&As[row][kb + 8 + 2 * p];
            }
        }
        #pragma unroll
        for (int j = 0; j < 4; ++j) {
            const int col = wn * 64 + j * 16 + l16;
            const int kb2 = half * 16;  // lanes 0-15: K 0..15 ; lanes 16-31: K 16..31
            #pragma unroll
            for (int p = 0; p < 8; ++p)
                bfr[j].u[p] = *(const unsigned int*)&Bs[col][kb2 + 2 * p];
        }
        #pragma unroll
        for (int i = 0; i < 2; ++i)
            #pragma unroll
            for (int j = 0; j < 4; ++j)
                acc[i][j] = __builtin_amdgcn_wmma_f32_16x16x32_bf16(
                    false, af[i].v, false, bfr[j].v, (short)0, acc[i][j], false, false);
        __syncthreads();
    }

    #pragma unroll
    for (int i = 0; i < 2; ++i)
        #pragma unroll
        for (int j = 0; j < 4; ++j)
            #pragma unroll
            for (int r = 0; r < 8; ++r) {
                int row = tM + wm * 32 + i * 16 + r + half * 8;
                int col = tN + wn * 64 + j * 16 + l16;
                if (row < M && col < N) {
                    float v = acc[i][j][r];
                    if (bias) v += bias[col];
                    if (act)  v = 0.5f * v * (1.f + erff(v * 0.70710678118654752f));
                    if (res)  v += res[(size_t)row * N + col];
                    C[(size_t)row * N + col] = v;
                }
            }
}

// ---------------------------------------------------------------------------
// Fused BiFormer attention: one wave per (b, region, head).
// q[16x64] x gathered K[64x64]^T -> softmax -> x V[64x64]. 16 WMMAs per block.
// qkv [B,208,2304] with pad rows zeroed. out compacted [B,197,768].
// ---------------------------------------------------------------------------
#define ALD 68   // LDS half-stride: rows 136B -> 8B aligned packed stores

__global__ __launch_bounds__(32)
void biformer_attn_kernel(const float* __restrict__ qkv, const int* __restrict__ idx,
                          float* __restrict__ out)
{
    __shared__ unsigned short qs[16][ALD];
    __shared__ unsigned short ks[64][ALD];   // gathered K rows: ks[n][d]
    __shared__ unsigned short vt[64][ALD];   // V transposed:   vt[d][n]
    __shared__ float          att[16][64];
    __shared__ unsigned short ab[16][ALD];   // softmaxed attention, bf16

    const int x    = blockIdx.x;
    const int hh   = x % 12;
    const int r    = (x / 12) % 13;
    const int b    = x / 156;
    const int lane = threadIdx.x;
    const int half = lane >> 4;
    const int l16  = lane & 15;

    for (int i = lane; i < 16 * 16; i += 32) {
        int m = i >> 4, dq = (i & 15) << 2;
        const float4 qv = *(const float4*)(qkv +
            ((size_t)(b * 208 + r * 16 + m)) * 2304 + hh * 64 + dq);
        *(uint2*)&qs[m][dq] = make_uint2(pk2(qv.x, qv.y), pk2(qv.z, qv.w));
    }
    for (int i = lane; i < 64 * 16; i += 32) {
        int n = i >> 4, dq = (i & 15) << 2;
        int g = idx[(b * 13 + r) * 4 + (n >> 4)];
        size_t base = ((size_t)(b * 208 + g * 16 + (n & 15))) * 2304 + hh * 64 + dq;
        const float4 kv = *(const float4*)(qkv + base + 768);
        const float4 vv = *(const float4*)(qkv + base + 1536);
        *(uint2*)&ks[n][dq] = make_uint2(pk2(kv.x, kv.y), pk2(kv.z, kv.w));
        vt[dq + 0][n] = f2bf(vv.x);
        vt[dq + 1][n] = f2bf(vv.y);
        vt[dq + 2][n] = f2bf(vv.z);
        vt[dq + 3][n] = f2bf(vv.w);
    }
    __syncthreads();

    // att = q . K^T   (M=16, N=64 keys, K=64 channels)
    v8f s[4] = {};
    #pragma unroll
    for (int k0 = 0; k0 < 64; k0 += 32) {
        Frag16 a;
        const int kb = half * 16;
        #pragma unroll
        for (int p = 0; p < 4; ++p) {
            a.u[p]     = *(const unsigned int*)&qs[l16][k0 + kb + 2 * p];
            a.u[4 + p] = *(const unsigned int*)&qs[l16][k0 + kb + 8 + 2 * p];
        }
        #pragma unroll
        for (int t = 0; t < 4; ++t) {
            Frag16 bb;
            const int kb2 = half * 16;
            #pragma unroll
            for (int p = 0; p < 8; ++p)
                bb.u[p] = *(const unsigned int*)&ks[t * 16 + l16][k0 + kb2 + 2 * p];
            s[t] = __builtin_amdgcn_wmma_f32_16x16x32_bf16(
                false, a.v, false, bb.v, (short)0, s[t], false, false);
        }
    }
    #pragma unroll
    for (int t = 0; t < 4; ++t)
        #pragma unroll
        for (int rr = 0; rr < 8; ++rr)
            att[rr + half * 8][t * 16 + l16] = s[t][rr] * 0.125f;  // dh^-0.5
    __syncthreads();

    if (lane < 16) {
        float mx = -1e30f;
        for (int n = 0; n < 64; ++n) mx = fmaxf(mx, att[lane][n]);
        float sum = 0.f;
        for (int n = 0; n < 64; ++n) sum += __expf(att[lane][n] - mx);
        float inv = 1.f / sum;
        for (int n = 0; n < 64; ++n)
            ab[lane][n] = f2bf(__expf(att[lane][n] - mx) * inv);
    }
    __syncthreads();

    // out = att . V   (M=16, N=64 channels, K=64 keys)
    v8f o[4] = {};
    #pragma unroll
    for (int k0 = 0; k0 < 64; k0 += 32) {
        Frag16 a;
        const int kb = half * 16;
        #pragma unroll
        for (int p = 0; p < 4; ++p) {
            a.u[p]     = *(const unsigned int*)&ab[l16][k0 + kb + 2 * p];
            a.u[4 + p] = *(const unsigned int*)&ab[l16][k0 + kb + 8 + 2 * p];
        }
        #pragma unroll
        for (int t = 0; t < 4; ++t) {
            Frag16 bb;
            const int kb2 = half * 16;
            #pragma unroll
            for (int p = 0; p < 8; ++p)
                bb.u[p] = *(const unsigned int*)&vt[t * 16 + l16][k0 + kb2 + 2 * p];
            o[t] = __builtin_amdgcn_wmma_f32_16x16x32_bf16(
                false, a.v, false, bb.v, (short)0, o[t], false, false);
        }
    }
    #pragma unroll
    for (int t = 0; t < 4; ++t)
        #pragma unroll
        for (int rr = 0; rr < 8; ++rr) {
            int m = rr + half * 8;
            int tok = r * 16 + m;
            if (tok < 197)
                out[((size_t)(b * 197 + tok)) * 768 + hh * 64 + t * 16 + l16] = o[t][rr];
        }
}

// ---------------------------------------------------------------------------
// LayerNorm over C=768 per token. Block per token row, 256 threads x 3 ch.
// ---------------------------------------------------------------------------
__global__ __launch_bounds__(256)
void layernorm_kernel(const float* __restrict__ in, const float* __restrict__ g,
                      const float* __restrict__ beta, float* __restrict__ out,
                      int tokens_in, int tokens_out)
{
    __shared__ float red[256];
    const int row = blockIdx.x;
    const int b   = row / tokens_in;
    const int tok = row % tokens_in;
    const int t0  = threadIdx.x;
    const float* xi = in + (size_t)row * 768;

    float v0 = xi[t0], v1 = xi[t0 + 256], v2 = xi[t0 + 512];
    red[t0] = v0 + v1 + v2;
    __syncthreads();
    for (int s = 128; s > 0; s >>= 1) {
        if (t0 < s) red[t0] += red[t0 + s];
        __syncthreads();
    }
    float mu = red[0] * (1.f / 768.f);
    __syncthreads();
    float d0 = v0 - mu, d1 = v1 - mu, d2 = v2 - mu;
    red[t0] = d0 * d0 + d1 * d1 + d2 * d2;
    __syncthreads();
    for (int s = 128; s > 0; s >>= 1) {
        if (t0 < s) red[t0] += red[t0 + s];
        __syncthreads();
    }
    float rinv = rsqrtf(red[0] * (1.f / 768.f) + 1e-5f);
    float* yo = out + ((size_t)b * tokens_out + tok) * 768;
    yo[t0]       = d0 * rinv * g[t0]       + beta[t0];
    yo[t0 + 256] = d1 * rinv * g[t0 + 256] + beta[t0 + 256];
    yo[t0 + 512] = d2 * rinv * g[t0 + 512] + beta[t0 + 512];
}

// Zero pad-token rows (197..207) of a [16,208,C] buffer.
__global__ void zero_pad_kernel(float* __restrict__ buf, int C)
{
    int i = blockIdx.x * 256 + threadIdx.x;
    int total = 16 * 11 * C;
    if (i < total) {
        int b = i / (11 * C);
        int rem = i % (11 * C);
        int row = 197 + rem / C;
        int c = rem % C;
        buf[((size_t)(b * 208 + row)) * C + c] = 0.f;
    }
}

// Region means of q and k channels from qkv [B,208,2304].
__global__ __launch_bounds__(256)
void region_mean_kernel(const float* __restrict__ qkv,
                        float* __restrict__ qreg, float* __restrict__ kreg)
{
    const int br = blockIdx.x;         // b*13 + r
    const int b = br / 13, r = br % 13;
    for (int c = threadIdx.x; c < 768; c += 256) {
        float sq = 0.f, sk = 0.f;
        for (int m = 0; m < 16; ++m) {
            size_t base = ((size_t)(b * 208 + r * 16 + m)) * 2304 + c;
            sq += qkv[base];
            sk += qkv[base + 768];
        }
        qreg[(size_t)br * 768 + c] = sq * (1.f / 16.f);
        kreg[(size_t)br * 768 + c] = sk * (1.f / 16.f);
    }
}

// Region affinity + top-4 selection. One 32-thread block per (b,r).
__global__ __launch_bounds__(32)
void topk_kernel(const float* __restrict__ qreg, const float* __restrict__ kreg,
                 int* __restrict__ idx)
{
    __shared__ float aff[16];
    const int br = blockIdx.x;
    const int b  = br / 13;
    const int s  = threadIdx.x;
    if (s < 13) {
        const float* qv = qreg + (size_t)br * 768;
        const float* kv = kreg + ((size_t)(b * 13 + s)) * 768;
        float d = 0.f;
        for (int c = 0; c < 768; ++c) d += qv[c] * kv[c];
        aff[s] = d;
    }
    __syncthreads();
    if (threadIdx.x == 0) {
        bool taken[13] = {};
        for (int j = 0; j < 4; ++j) {
            int best = 0; float bv = -3.4e38f;
            for (int q = 0; q < 13; ++q)
                if (!taken[q] && aff[q] > bv) { bv = aff[q]; best = q; }
            taken[best] = true;
            idx[br * 4 + j] = best;  // strict '>' => lowest index on ties (jax top_k)
        }
    }
}

// im2col for the 16x16/stride-16 patch conv. col index = ci*256 + ky*16 + kx.
__global__ __launch_bounds__(256)
void im2col_kernel(const float* __restrict__ x, float* __restrict__ pc)
{
    const int row = blockIdx.x;        // b*196 + n
    const int b = row / 196, n = row % 196;
    const int py = n / 14, px = n % 14;
    for (int c = threadIdx.x; c < 768; c += 256) {
        int ci = c >> 8, rem = c & 255, ky = rem >> 4, kx = rem & 15;
        pc[(size_t)row * 768 + c] =
            x[(((size_t)(b * 3 + ci)) * 224 + py * 16 + ky) * 224 + px * 16 + kx];
    }
}

// conv_w [768(out), 768(in*kh*kw)] -> Wp [768(k), 768(out)]
__global__ void wtrans_kernel(const float* __restrict__ w, float* __restrict__ wt)
{
    int i = blockIdx.x * 256 + threadIdx.x;
    if (i < 768 * 768) {
        int o = i / 768, k = i % 768;
        wt[(size_t)k * 768 + o] = w[i];
    }
}

// Assemble tokens: cls + patches, add pos_embed. t [B,197,768].
__global__ __launch_bounds__(256)
void assemble_kernel(const float* __restrict__ pout, const float* __restrict__ cls,
                     const float* __restrict__ pos, float* __restrict__ t)
{
    const int row = blockIdx.x;        // b*197 + tok
    const int b = row / 197, tok = row % 197;
    for (int c = threadIdx.x; c < 768; c += 256) {
        float v = (tok == 0) ? cls[c]
                             : pout[((size_t)(b * 196 + tok - 1)) * 768 + c];
        t[(size_t)row * 768 + c] = v + pos[(size_t)tok * 768 + c];
    }
}

// Gather cls rows into a contiguous [16,768] matrix for the head GEMM.
__global__ void clscopy_kernel(const float* __restrict__ t, float* __restrict__ cb)
{
    int i = blockIdx.x * 256 + threadIdx.x;
    if (i < 16 * 768) {
        int b = i / 768, c = i % 768;
        cb[i] = t[((size_t)b * 197) * 768 + c];
    }
}

// ---------------------------------------------------------------------------
extern "C" void kernel_launch(void* const* d_in, const int* in_sizes, int n_in,
                              void* d_out, int out_size, void* d_ws, size_t ws_size,
                              hipStream_t stream) {
    (void)in_sizes; (void)n_in; (void)out_size; (void)ws_size;
    const float* x      = (const float*)d_in[0];
    const float* conv_w = (const float*)d_in[1];
    const float* conv_b = (const float*)d_in[2];
    const float* cls    = (const float*)d_in[3];
    const float* pos    = (const float*)d_in[4];
    const float* ln1_g  = (const float*)d_in[5];
    const float* ln1_b  = (const float*)d_in[6];
    const float* qkv_w  = (const float*)d_in[7];
    const float* qkv_b  = (const float*)d_in[8];
    const float* proj_w = (const float*)d_in[9];
    const float* proj_b = (const float*)d_in[10];
    const float* ln2_g  = (const float*)d_in[11];
    const float* ln2_b  = (const float*)d_in[12];
    const float* fc1_w  = (const float*)d_in[13];
    const float* fc1_b  = (const float*)d_in[14];
    const float* fc2_w  = (const float*)d_in[15];
    const float* fc2_b  = (const float*)d_in[16];
    const float* head_w = (const float*)d_in[17];
    const float* head_b = (const float*)d_in[18];
    float* out = (float*)d_out;

    float* W = (float*)d_ws;
    size_t o = 0;
    float* t_   = W + o; o += (size_t)16 * 197 * 768;    // residual stream
    float* h_   = W + o; o += (size_t)16 * 208 * 768;    // LN1 out, padded layout
    float* qkv_ = W + o; o += (size_t)16 * 208 * 2304;   // qkv, padded layout
    float* ao   = W + o; o += (size_t)16 * 197 * 768;    // attn out / LN2 out
    float* p1   = W + o; o += (size_t)16 * 197 * 768;    // proj out
    float* m1   = W + o; o += (size_t)16 * 197 * 3072;   // fc1 out
    float* qreg = W + o; o += (size_t)16 * 13 * 768;
    float* kreg = W + o; o += (size_t)16 * 13 * 768;
    int*   idx  = (int*)(W + o); o += (size_t)16 * 13 * 4;
    float* pcol = W + o; o += (size_t)16 * 196 * 768;
    float* pout = W + o; o += (size_t)16 * 196 * 768;
    float* wp   = W + o; o += (size_t)768 * 768;
    float* cb   = W + o; o += (size_t)16 * 768;

    auto gemm = [&](const float* A, const float* Bm, const float* bias,
                    const float* res, float* C, int M, int N, int K, int act) {
        dim3 grid((N + BN - 1) / BN, (M + BM - 1) / BM);
        gemm_bf16_wmma<<<grid, dim3(256), 0, stream>>>(A, Bm, bias, res, C, M, N, K, act);
    };

    // pad rows of h_ stay zero across all blocks (LN only writes real rows)
    zero_pad_kernel<<<(16 * 11 * 768 + 255) / 256, 256, 0, stream>>>(h_, 768);

    // patch embedding as GEMM
    im2col_kernel<<<16 * 196, 256, 0, stream>>>(x, pcol);
    wtrans_kernel<<<(768 * 768 + 255) / 256, 256, 0, stream>>>(conv_w, wp);
    gemm(pcol, wp, conv_b, nullptr, pout, 16 * 196, 768, 768, 0);
    assemble_kernel<<<16 * 197, 256, 0, stream>>>(pout, cls, pos, t_);

    for (int l = 0; l < 12; ++l) {
        layernorm_kernel<<<16 * 197, 256, 0, stream>>>(
            t_, ln1_g + (size_t)l * 768, ln1_b + (size_t)l * 768, h_, 197, 208);
        gemm(h_, qkv_w + (size_t)l * 768 * 2304, qkv_b + (size_t)l * 2304,
             nullptr, qkv_, 16 * 208, 2304, 768, 0);
        // reference pads qkv AFTER bias add with zeros -> force pad rows to 0
        zero_pad_kernel<<<(16 * 11 * 2304 + 255) / 256, 256, 0, stream>>>(qkv_, 2304);
        region_mean_kernel<<<16 * 13, 256, 0, stream>>>(qkv_, qreg, kreg);
        topk_kernel<<<16 * 13, 32, 0, stream>>>(qreg, kreg, idx);
        biformer_attn_kernel<<<16 * 13 * 12, 32, 0, stream>>>(qkv_, idx, ao);
        gemm(ao, proj_w + (size_t)l * 768 * 768, proj_b + (size_t)l * 768,
             nullptr, p1, 16 * 197, 768, 768, 0);
        layernorm_kernel<<<16 * 197, 256, 0, stream>>>(
            p1, ln2_g + (size_t)l * 768, ln2_b + (size_t)l * 768, ao, 197, 197);
        gemm(ao, fc1_w + (size_t)l * 768 * 3072, fc1_b + (size_t)l * 3072,
             nullptr, m1, 16 * 197, 3072, 768, 1);
        gemm(m1, fc2_w + (size_t)l * 3072 * 768, fc2_b + (size_t)l * 768,
             t_, t_, 16 * 197, 768, 3072, 0);   // residual: t += mlp(...)
    }

    clscopy_kernel<<<(16 * 768 + 255) / 256, 256, 0, stream>>>(t_, cb);
    gemm(cb, head_w, head_b, nullptr, out, 16, 1000, 768, 0);
}
